// SmolVLMTextAttention_42236708389215
// MI455X (gfx1250) — compile-verified, matrix operand order checked by reading, not run on hardware
//
#include <hip/hip_runtime.h>
#include <hip/hip_bf16.h>

// ---------------------------------------------------------------------------
// SmolVLM text attention forward for MI455X (gfx1250, wave32, WMMA bf16).
//   B=2, S=2048, D=2048, H=32, KVH=8, DH=64, G=4
// Pipeline:
//   1) fp32 -> bf16 convert (hidden + weights)
//   2) QKV GEMM (32x64 tile/wave, v_wmma_f32_16x16x32_bf16) + fused RoPE,
//      V stored transposed (dh-major) for the PV GEMM
//   3) transposed flash attention: S^T = K*Q^T, per-lane online softmax,
//      O^T = V^T * P^T (P^T fragment assembled directly from score regs)
//   4) output GEMM (32x64 tile/wave) -> fp32 d_out
// ---------------------------------------------------------------------------

#define B_   2
#define S_   2048
#define D_   2048
#define H_   32
#define KVH_ 8
#define DH_  64
#define G_   4
#define SCALE_ 0.125f
#define NEG_ (-1.0e9f)

typedef __bf16 bf16;
typedef __attribute__((ext_vector_type(16))) __bf16 v16bf;
typedef __attribute__((ext_vector_type(8)))  __bf16 v8bf;
typedef __attribute__((ext_vector_type(8)))  float  v8f;

// Load a 16x32 bf16 WMMA operand fragment (A pattern; also valid B pattern
// when the logical B column is contiguous in memory, i.e. K-major source).
//   lanes 0-15 : K {0..7} in elems 0-7,  K {16..23} in elems 8-15
//   lanes 16-31: K {8..15} in elems 0-7, K {24..31} in elems 8-15
__device__ __forceinline__ v16bf load_frag(const bf16* __restrict__ base,
                                           int stride, int lane) {
  const bf16* p = base + (size_t)(lane & 15) * stride + ((lane >> 4) << 3);
  v8bf lo = *(const v8bf*)(p);
  v8bf hi = *(const v8bf*)(p + 16);
  v16bf r;
#pragma unroll
  for (int i = 0; i < 8; ++i) { r[i] = lo[i]; r[i + 8] = hi[i]; }
  return r;
}

__device__ __forceinline__ v8f wmma_bf16(v16bf a, v16bf b, v8f c) {
  return __builtin_amdgcn_wmma_f32_16x16x32_bf16(
      /*neg_a=*/false, a, /*neg_b=*/false, b,
      /*c_mod=*/(short)0, c, /*reuse_a=*/false, /*reuse_b=*/false);
}

// ---------------------------------------------------------------------------
// fp32 -> bf16 conversion (all sizes divisible by 4)
// ---------------------------------------------------------------------------
__global__ void cvt_f32_bf16(const float* __restrict__ in,
                             bf16* __restrict__ out, int n) {
  int i = blockIdx.x * blockDim.x + threadIdx.x;
  int idx = i * 4;
  if (idx + 3 < n) {
    float4 v = *(const float4*)(in + idx);
    out[idx + 0] = (bf16)v.x;
    out[idx + 1] = (bf16)v.y;
    out[idx + 2] = (bf16)v.z;
    out[idx + 3] = (bf16)v.w;
  }
}

// ---------------------------------------------------------------------------
// QKV projection + RoPE.  One wave: 32 rows x one head (N=64).
// 8 WMMAs per k-step for 6 fragment loads (B frags shared across A tiles).
// blockDim=128 (4 waves). grid=(128/4, H+2*KVH).
// ---------------------------------------------------------------------------
__global__ void qkv_rope_kernel(const bf16* __restrict__ X,
                                const bf16* __restrict__ Wq,
                                const bf16* __restrict__ Wk,
                                const bf16* __restrict__ Wv,
                                const float* __restrict__ cosp,
                                const float* __restrict__ sinp,
                                bf16* __restrict__ Qw,
                                bf16* __restrict__ Kw,
                                bf16* __restrict__ VTw) {
  const int lane = threadIdx.x & 31;
  const int wid  = threadIdx.x >> 5;
  const int wt   = blockIdx.x * 4 + wid;   // 0..127 (32-row tasks over B*S)
  const int hh   = blockIdx.y;             // 0..H+2*KVH-1
  const int m0   = wt * 32;
  const int b    = m0 / S_;
  const int s0   = m0 % S_;

  const bf16* W;
  int headrow, mode;                        // 0=Q 1=K 2=V
  if (hh < H_)              { W = Wq; headrow = hh * DH_;                mode = 0; }
  else if (hh < H_ + KVH_)  { W = Wk; headrow = (hh - H_) * DH_;         mode = 1; }
  else                      { W = Wv; headrow = (hh - H_ - KVH_) * DH_;  mode = 2; }

  const bf16* arow0 = X + (size_t)m0 * D_;
  const bf16* arow1 = arow0 + (size_t)16 * D_;
  const bf16* brow  = W + (size_t)headrow * D_;

  v8f c00 = {}, c01 = {}, c02 = {}, c03 = {};
  v8f c10 = {}, c11 = {}, c12 = {}, c13 = {};
  for (int k0 = 0; k0 < D_; k0 += 32) {
    v16bf a0 = load_frag(arow0 + k0, D_, lane);
    v16bf a1 = load_frag(arow1 + k0, D_, lane);
    v16bf b0 = load_frag(brow + k0,           D_, lane);
    v16bf b1 = load_frag(brow + 16 * D_ + k0, D_, lane);
    v16bf b2 = load_frag(brow + 32 * D_ + k0, D_, lane);
    v16bf b3 = load_frag(brow + 48 * D_ + k0, D_, lane);
    c00 = wmma_bf16(a0, b0, c00);  c10 = wmma_bf16(a1, b0, c10);
    c01 = wmma_bf16(a0, b1, c01);  c11 = wmma_bf16(a1, b1, c11);
    c02 = wmma_bf16(a0, b2, c02);  c12 = wmma_bf16(a1, b2, c12);
    c03 = wmma_bf16(a0, b3, c03);  c13 = wmma_bf16(a1, b3, c13);
  }

  const int ln = lane & 15;
  const int rg = (lane >> 4) << 3;          // +8 row offset for upper lanes

  if (mode == 2) {
    // store V transposed: VT[b][kvh][dh][s]
    const int kvh = hh - H_ - KVH_;
    const size_t vb = ((size_t)(b * KVH_ + kvh)) * DH_;
#pragma unroll
    for (int mi = 0; mi < 2; ++mi) {
      const v8f& t0 = mi ? c10 : c00;
      const v8f& t1 = mi ? c11 : c01;
      const v8f& t2 = mi ? c12 : c02;
      const v8f& t3 = mi ? c13 : c03;
      int sm = s0 + mi * 16;
#pragma unroll
      for (int r = 0; r < 8; ++r) {
        int s = sm + r + rg;
        VTw[(vb + 0  + ln) * S_ + s] = (bf16)t0[r];
        VTw[(vb + 16 + ln) * S_ + s] = (bf16)t1[r];
        VTw[(vb + 32 + ln) * S_ + s] = (bf16)t2[r];
        VTw[(vb + 48 + ln) * S_ + s] = (bf16)t3[r];
      }
    }
    return;
  }

  // RoPE: rotate_half pairs dh <-> dh+32, i.e. tiles (0,2) and (1,3).
  // cos/sin repeat: cos[dh+32] == cos[dh].
#pragma unroll
  for (int mi = 0; mi < 2; ++mi) {
    int sm = s0 + mi * 16;
#pragma unroll
    for (int pg = 0; pg < 2; ++pg) {
      const v8f& x1v = mi ? (pg ? c11 : c10) : (pg ? c01 : c00);
      const v8f& x2v = mi ? (pg ? c13 : c12) : (pg ? c03 : c02);
      const int dh = pg * 16 + ln;          // < 32
#pragma unroll
      for (int r = 0; r < 8; ++r) {
        int s = sm + r + rg;
        float cv = cosp[((size_t)b * S_ + s) * DH_ + dh];
        float sv = sinp[((size_t)b * S_ + s) * DH_ + dh];
        float x1 = x1v[r], x2 = x2v[r];
        float o1 = x1 * cv - x2 * sv;
        float o2 = x2 * cv + x1 * sv;
        if (mode == 0) {
          o1 *= SCALE_; o2 *= SCALE_;
          size_t qb = (((size_t)b * H_ + hh) * S_ + s) * DH_;
          Qw[qb + dh]      = (bf16)o1;
          Qw[qb + dh + 32] = (bf16)o2;
        } else {
          int kvh = hh - H_;
          size_t kb = (((size_t)b * KVH_ + kvh) * S_ + s) * DH_;
          Kw[kb + dh]      = (bf16)o1;
          Kw[kb + dh + 32] = (bf16)o2;
        }
      }
    }
  }
}

// ---------------------------------------------------------------------------
// Transposed causal flash attention. One wave: 16 queries of one (b,h).
// S^T = K * Q^T  (M=keys across VGPRs, N=queries across lanes)
// O^T = V^T * P^T (M=dh, N=queries) -> per-query stats live per lane.
// blockDim=128. grid=(S/16/4, B*H).
// ---------------------------------------------------------------------------
__global__ void attn_kernel(const bf16* __restrict__ Qw,
                            const bf16* __restrict__ Kw,
                            const bf16* __restrict__ VTw,
                            bf16* __restrict__ AT) {
  const int lane = threadIdx.x & 31;
  const int wid  = threadIdx.x >> 5;
  const int qt   = blockIdx.x * 4 + wid;    // 0..127
  const int bh   = blockIdx.y;              // 0..63
  const int b    = bh >> 5;
  const int h    = bh & 31;
  const int kvh  = h >> 2;                  // G=4
  const int q0   = qt * 16;
  const int ln   = lane & 15;
  const int rg   = (lane >> 4) << 3;

  const bf16* qbase = Qw  + (((size_t)b * H_  + h)    * S_ + q0) * DH_;
  const bf16* kbase = Kw  + (((size_t)b * KVH_ + kvh) * S_) * DH_;
  const bf16* vbase = VTw + (((size_t)b * KVH_ + kvh) * DH_) * S_;

  // Q^T B-fragments (reused over all key blocks): lane = query column.
  v16bf qb0 = load_frag(qbase,      DH_, lane);   // dh 0..31
  v16bf qb1 = load_frag(qbase + 32, DH_, lane);   // dh 32..63

  v8f o0 = {}, o1 = {}, o2 = {}, o3 = {};
  float m_run = -1e30f, l_run = 0.0f;
  const int qmax = q0 + 15;
  const int qcol = q0 + ln;                 // this lane's query index

  for (int kb = 0; kb <= qmax; kb += 32) {
    // ---- scores^T: two 16-key tiles ----
    v16bf ka0 = load_frag(kbase + (size_t)kb * DH_,             DH_, lane);
    v16bf ka1 = load_frag(kbase + (size_t)kb * DH_ + 32,        DH_, lane);
    v16bf kc0 = load_frag(kbase + (size_t)(kb + 16) * DH_,      DH_, lane);
    v16bf kc1 = load_frag(kbase + (size_t)(kb + 16) * DH_ + 32, DH_, lane);
    v8f s0 = {}, s1 = {};
    s0 = wmma_bf16(ka0, qb0, s0);
    s0 = wmma_bf16(ka1, qb1, s0);
    s1 = wmma_bf16(kc0, qb0, s1);
    s1 = wmma_bf16(kc1, qb1, s1);

    // ---- causal mask (key > query -> -1e9) ----
#pragma unroll
    for (int r = 0; r < 8; ++r) {
      int key0 = kb + r + rg;
      int key1 = key0 + 16;
      s0[r] = (key0 > qcol) ? NEG_ : s0[r];
      s1[r] = (key1 > qcol) ? NEG_ : s1[r];
    }

    // ---- online softmax, per-query = per-lane(+xor16) ----
    float mb = -1e30f;
#pragma unroll
    for (int r = 0; r < 8; ++r) mb = fmaxf(mb, fmaxf(s0[r], s1[r]));
    mb = fmaxf(mb, __shfl_xor(mb, 16, 32));
    float m_new = fmaxf(m_run, mb);
    float alpha = __expf(m_run - m_new);

    float lb = 0.0f;
    v16bf pb;                                // P^T B-fragment, built in-regs
#pragma unroll
    for (int r = 0; r < 8; ++r) {
      float p0 = __expf(s0[r] - m_new);
      float p1 = __expf(s1[r] - m_new);
      lb += p0 + p1;
      pb[r]     = (bf16)p0;                  // keys kb+{0..7}/{8..15}
      pb[r + 8] = (bf16)p1;                  // keys kb+{16..23}/{24..31}
    }
    lb += __shfl_xor(lb, 16, 32);
    l_run = l_run * alpha + lb;
    m_run = m_new;

#pragma unroll
    for (int r = 0; r < 8; ++r) {
      o0[r] *= alpha; o1[r] *= alpha; o2[r] *= alpha; o3[r] *= alpha;
    }

    // ---- O^T += V^T * P^T ----
    v16bf va0 = load_frag(vbase + (size_t)0  * S_ + kb, S_, lane);
    v16bf va1 = load_frag(vbase + (size_t)16 * S_ + kb, S_, lane);
    v16bf va2 = load_frag(vbase + (size_t)32 * S_ + kb, S_, lane);
    v16bf va3 = load_frag(vbase + (size_t)48 * S_ + kb, S_, lane);
    o0 = wmma_bf16(va0, pb, o0);
    o1 = wmma_bf16(va1, pb, o1);
    o2 = wmma_bf16(va2, pb, o2);
    o3 = wmma_bf16(va3, pb, o3);
  }

  const float inv_l = 1.0f / l_run;
  // O^T element (tile t, vgpr r, lane): dh = t*16 + r + rg, query = q0+ln
  const int q = q0 + ln;
  size_t abase = ((size_t)(b * S_ + q)) * (H_ * DH_) + (size_t)h * DH_;
#pragma unroll
  for (int r = 0; r < 8; ++r) {
    AT[abase + 0  + r + rg] = (bf16)(o0[r] * inv_l);
    AT[abase + 16 + r + rg] = (bf16)(o1[r] * inv_l);
    AT[abase + 32 + r + rg] = (bf16)(o2[r] * inv_l);
    AT[abase + 48 + r + rg] = (bf16)(o3[r] * inv_l);
  }
}

// ---------------------------------------------------------------------------
// Output projection: out = attn @ Wo^T (fp32 out).
// One wave: 32 rows x 64 cols. blockDim=128. grid=(128/4, D/64).
// ---------------------------------------------------------------------------
__global__ void out_gemm_kernel(const bf16* __restrict__ AT,
                                const bf16* __restrict__ Wo,
                                float* __restrict__ out) {
  const int lane = threadIdx.x & 31;
  const int wid  = threadIdx.x >> 5;
  const int wt   = blockIdx.x * 4 + wid;    // 0..127
  const int n0   = blockIdx.y * 64;         // 0..2047
  const int m0   = wt * 32;

  const bf16* arow0 = AT + (size_t)m0 * D_;
  const bf16* arow1 = arow0 + (size_t)16 * D_;
  const bf16* brow  = Wo + (size_t)n0 * D_;

  v8f c00 = {}, c01 = {}, c02 = {}, c03 = {};
  v8f c10 = {}, c11 = {}, c12 = {}, c13 = {};
  for (int k0 = 0; k0 < D_; k0 += 32) {
    v16bf a0 = load_frag(arow0 + k0, D_, lane);
    v16bf a1 = load_frag(arow1 + k0, D_, lane);
    v16bf b0 = load_frag(brow + k0,           D_, lane);
    v16bf b1 = load_frag(brow + 16 * D_ + k0, D_, lane);
    v16bf b2 = load_frag(brow + 32 * D_ + k0, D_, lane);
    v16bf b3 = load_frag(brow + 48 * D_ + k0, D_, lane);
    c00 = wmma_bf16(a0, b0, c00);  c10 = wmma_bf16(a1, b0, c10);
    c01 = wmma_bf16(a0, b1, c01);  c11 = wmma_bf16(a1, b1, c11);
    c02 = wmma_bf16(a0, b2, c02);  c12 = wmma_bf16(a1, b2, c12);
    c03 = wmma_bf16(a0, b3, c03);  c13 = wmma_bf16(a1, b3, c13);
  }

  const int ln = lane & 15;
  const int rg = (lane >> 4) << 3;
#pragma unroll
  for (int mi = 0; mi < 2; ++mi) {
    const v8f& t0 = mi ? c10 : c00;
    const v8f& t1 = mi ? c11 : c01;
    const v8f& t2 = mi ? c12 : c02;
    const v8f& t3 = mi ? c13 : c03;
#pragma unroll
    for (int r = 0; r < 8; ++r) {
      size_t row = (size_t)(m0 + mi * 16 + r + rg) * D_;
      out[row + n0 + 0  + ln] = t0[r];
      out[row + n0 + 16 + ln] = t1[r];
      out[row + n0 + 32 + ln] = t2[r];
      out[row + n0 + 48 + ln] = t3[r];
    }
  }
}

// ---------------------------------------------------------------------------
extern "C" void kernel_launch(void* const* d_in, const int* in_sizes, int n_in,
                              void* d_out, int out_size, void* d_ws, size_t ws_size,
                              hipStream_t stream) {
  const float* hs   = (const float*)d_in[0];   // (B,S,D)
  const float* cosp = (const float*)d_in[1];   // (B,S,DH)
  const float* sinp = (const float*)d_in[2];   // (B,S,DH)
  // d_in[3] = attention_mask: causal, computed inline instead
  const float* Wq   = (const float*)d_in[4];   // (H*DH, D)
  const float* Wk   = (const float*)d_in[5];   // (KVH*DH, D)
  const float* Wv   = (const float*)d_in[6];   // (KVH*DH, D)
  const float* Wo   = (const float*)d_in[7];   // (D, H*DH)
  float* outp = (float*)d_out;                 // (B,S,D) fp32

  // workspace layout (bf16 elements)
  const size_t nX  = (size_t)B_ * S_ * D_;         // 8388608
  const size_t nWq = (size_t)H_ * DH_ * D_;        // 4194304
  const size_t nWk = (size_t)KVH_ * DH_ * D_;      // 1048576
  const size_t nQ  = (size_t)B_ * H_ * S_ * DH_;   // 8388608
  const size_t nK  = (size_t)B_ * KVH_ * S_ * DH_; // 2097152

  bf16* Xb  = (bf16*)d_ws;
  bf16* Wqb = Xb  + nX;
  bf16* Wkb = Wqb + nWq;
  bf16* Wvb = Wkb + nWk;
  bf16* Wob = Wvb + nWk;
  bf16* Qw  = Wob + nWq;
  bf16* Kw  = Qw  + nQ;
  bf16* VTw = Kw  + nK;
  bf16* AT  = VTw + nK;       // (B,S,H*DH) bf16, nX elements

  const int thr = 256;
  auto cvt = [&](const float* src, bf16* dst, size_t n) {
    int blocks = (int)((n / 4 + thr - 1) / thr);
    cvt_f32_bf16<<<blocks, thr, 0, stream>>>(src, dst, (int)n);
  };
  cvt(hs, Xb,  nX);
  cvt(Wq, Wqb, nWq);
  cvt(Wk, Wkb, nWk);
  cvt(Wv, Wvb, nWk);
  cvt(Wo, Wob, nWq);

  // QKV + RoPE: 128 x 32-row tasks x 48 head-slots, 4 waves/block
  qkv_rope_kernel<<<dim3(32, H_ + 2 * KVH_), 128, 0, stream>>>(
      Xb, Wqb, Wkb, Wvb, cosp, sinp, Qw, Kw, VTw);

  // flash attention: 128 q-tiles x (B*H) heads, 4 waves/block
  attn_kernel<<<dim3(32, B_ * H_), 128, 0, stream>>>(Qw, Kw, VTw, AT);

  // output projection: 128 x 32-row tasks x 32 n-tiles
  out_gemm_kernel<<<dim3(32, D_ / 64), 128, 0, stream>>>(AT, Wob, outp);
}
